// EpisodeMultiheadAttentionBlock_89988154786371
// MI455X (gfx1250) — compile-verified, matrix operand order checked by reading, not run on hardware
//
#include <hip/hip_runtime.h>
#include <hip/hip_bf16.h>
#include <stdint.h>

// ---------------------------------------------------------------------------
// Problem constants (from reference): B=16, K_LEN=2048, Q_LEN=1024, E=512, H=8
// ---------------------------------------------------------------------------
#define B_DIM 16
#define K_SEQ 2048
#define Q_SEQ 1024
#define E_DIM 512
#define H_DIM 8
#define DH 64

typedef __attribute__((ext_vector_type(16))) _Float16 v16h;
typedef __attribute__((ext_vector_type(8)))  float    v8f;
typedef __attribute__((ext_vector_type(4)))  unsigned int u32x4;
typedef __attribute__((ext_vector_type(8)))  int      i32x8;
typedef __attribute__((ext_vector_type(4)))  int      i32x4;

#if __has_builtin(__builtin_amdgcn_tensor_load_to_lds) && \
    __has_builtin(__builtin_amdgcn_s_wait_tensorcnt)
#define USE_TDM 1
#else
#define USE_TDM 0
#endif

__device__ __forceinline__ v8f wmma16(v16h a, v16h b, v8f c) {
  // v_wmma_f32_16x16x32_f16: D = A(16x32 f16) * B(32x16 f16) + C(16x16 f32)
  return __builtin_amdgcn_wmma_f32_16x16x32_f16(false, a, false, b, (short)0, c,
                                                false, false);
}

union FragU { uint4 u[2]; v16h v; };

// Load one 16x32 f16 fragment row per the CDNA5 A/B VGPR layout:
// lane (l&15) = row; elems 0..7 = k (hi*8 + 0..7); elems 8..15 = k (16+hi*8+..)
__device__ __forceinline__ v16h load_frag16(const _Float16* row, int hi) {
  FragU f;
  f.u[0] = *(const uint4*)(row + hi * 8);
  f.u[1] = *(const uint4*)(row + 16 + hi * 8);
  return f.v;
}

#if USE_TDM
// ---------------------------------------------------------------------------
// TDM: async DMA of a 2D tile (32 halves x 64 rows, row stride 512 elems) from
// global into LDS, inserting 4 DWORDs of padding every 16 DWORDs so the LDS
// image matches a [64][40]-half padded tile (bank-conflict-free frag reads).
// D# layout per CDNA5 ISA ch.8 (group0: count/lds/global/type; group1: dims).
// ---------------------------------------------------------------------------
__device__ __forceinline__ void tdm_load_tile(uint32_t lds_off, uint64_t gaddr) {
  u32x4 g0;
  g0[0] = 1u;                                           // count=1, user mode
  g0[1] = lds_off;                                      // lds_addr (bytes)
  g0[2] = (uint32_t)gaddr;                              // global_addr[31:0]
  g0[3] = ((uint32_t)(gaddr >> 32) & 0x01FFFFFFu) | (2u << 30);  // addr hi, type=2
  i32x8 g1;
  g1[0] = (int)((1u << 16)        // data_size = 1 -> 2 bytes
              | (1u << 20)        // pad_enable
              | (3u << 22)        // pad_interval: 16 DWORDs (64B row)
              | (3u << 25));      // pad_amount: 4 DWORDs (16B pad)
  g1[1] = (int)(32u << 16);       // tensor_dim0 = 32 (bits 79:48, lo16)
  g1[2] = (int)(64u << 16);       // tensor_dim0 hi=0 | tensor_dim1 = 64 (lo16)
  g1[3] = (int)(32u << 16);       // tensor_dim1 hi=0 | tile_dim0 = 32
  g1[4] = (int)(64u);             // tile_dim1 = 64 | tile_dim2 = 0
  g1[5] = (int)(512u);            // tensor_dim0_stride = 512 elems (lo32)
  g1[6] = 0;                      // stride0 hi | stride1 lo
  g1[7] = 0;                      // stride1 hi
  i32x4 z = {0, 0, 0, 0};
#if defined(__clang_major__) && __clang_major__ >= 23
  i32x8 z8 = {};
  __builtin_amdgcn_tensor_load_to_lds(g0, g1, z, z, z8, 0);
#else
  __builtin_amdgcn_tensor_load_to_lds(g0, g1, z, z, 0);
#endif
}
#endif  // USE_TDM

// ---------------------------------------------------------------------------
// Elementwise helpers
// ---------------------------------------------------------------------------
__global__ void cast_f32_to_f16(const float* __restrict__ in,
                                _Float16* __restrict__ out, long long n) {
  long long i = (long long)blockIdx.x * blockDim.x + threadIdx.x;
  long long stride = (long long)gridDim.x * blockDim.x;
  for (; i < n; i += stride) out[i] = (_Float16)in[i];
}

__global__ void zero_f32_kernel(float* __restrict__ p, long long n) {
  long long i = (long long)blockIdx.x * blockDim.x + threadIdx.x;
  long long stride = (long long)gridDim.x * blockDim.x;
  for (; i < n; i += stride) p[i] = 0.0f;
}

// ---------------------------------------------------------------------------
// Generic WMMA GEMM: C[M,512] = act(A_f16[M,512] @ W_f16[N=512,512]^T + bias)
//                              (+ optional f32 residual)
// Macro tile 64x64, 256 threads = 8 waves, each wave: 2 m-tiles x 1 n-tile.
// Staging: TDM tensor_load_to_lds double-buffered (wave 0 issues DMA for tile
// s+1 while all waves compute tile s), fallback to per-thread b128 staging.
// A-row gather:    arow = (m/seqA)*strideA + baseA + m%seqA   (q = x[:, -Q:])
// outMode: 0 = f16 [M,512] ; 1 = f32 [M,512] ;
//          2 = f16 head-split  [b,h,s,dh]   (for Q,K)
//          3 = f16 head-split T [b,h,dh,s]  (for V: fragment-friendly)
// ---------------------------------------------------------------------------
__global__ __launch_bounds__(256)
void gemm_wmma_nt(const _Float16* __restrict__ A, const _Float16* __restrict__ W,
                  const float* __restrict__ bias, const float* __restrict__ resid,
                  float* __restrict__ out32, _Float16* __restrict__ out16,
                  int seqA, int strideA, int baseA,
                  int seqR, int strideR, int baseR,
                  int outMode, int seqOut, int doRelu) {
  __shared__ __align__(16) _Float16 As[2][64][40];  // 40-half row (80B, 16B mult)
  __shared__ __align__(16) _Float16 Bs[2][64][40];

  const int tid = threadIdx.x;
  const int wave = tid >> 5, lane = tid & 31;
  const int nlane = lane & 15, hi = lane >> 4;
  const int mt2 = wave >> 2;      // 0..1 -> m-tiles {mt2, mt2+2}
  const int nt  = wave & 3;       // n-tile
  const int m0 = blockIdx.x * 64, n0 = blockIdx.y * 64;

  // first row of this block's A strip (rows are contiguous: 64 | seqA always)
  const size_t arow0 = (size_t)(m0 / seqA) * strideA + baseA + (m0 % seqA);

  v8f acc0 = {};
  v8f acc1 = {};

#if USE_TDM
  const uint32_t lds_a = (uint32_t)(uintptr_t)(void*)&As[0][0][0];
  const uint32_t lds_b = (uint32_t)(uintptr_t)(void*)&Bs[0][0][0];
  const uint64_t ga = (uint64_t)(uintptr_t)(A + arow0 * E_DIM);
  const uint64_t gb = (uint64_t)(uintptr_t)(W + (size_t)n0 * E_DIM);
  const uint32_t bufBytes = 64u * 40u * 2u;  // 5120

  if (wave == 0) {  // prologue: DMA tile 0 into buffer 0
    tdm_load_tile(lds_a, ga);
    tdm_load_tile(lds_b, gb);
  }
  for (int s = 0; s < E_DIM / 32; ++s) {
    const int cur = s & 1;
    __syncthreads();  // everyone done computing on buf[1-cur] -> safe to refill
    if (wave == 0) {
      if (s + 1 < E_DIM / 32) {
        const uint32_t boff = (uint32_t)((s + 1) & 1) * bufBytes;
        const uint64_t koff = (uint64_t)(s + 1) * 32u * 2u;
        tdm_load_tile(lds_a + boff, ga + koff);
        tdm_load_tile(lds_b + boff, gb + koff);
        __builtin_amdgcn_s_wait_tensorcnt((short)2);  // current pair complete
      } else {
        __builtin_amdgcn_s_wait_tensorcnt((short)0);
      }
    }
    __syncthreads();  // publish buf[cur]
    v16h a0 = load_frag16(&As[cur][mt2 * 16 + nlane][0], hi);
    v16h a1 = load_frag16(&As[cur][(mt2 + 2) * 16 + nlane][0], hi);
    v16h bf = load_frag16(&Bs[cur][nt * 16 + nlane][0], hi);
    acc0 = wmma16(a0, bf, acc0);
    acc1 = wmma16(a1, bf, acc1);
  }
#else
  const int lr = tid >> 2;          // staging row 0..63
  const int lc = (tid & 3) * 8;     // staging col group (halves)
  const size_t arow = arow0 + 0;    // per-thread row = arow0 + lr (contiguous)
  for (int k0 = 0; k0 < E_DIM; k0 += 32) {
    __syncthreads();
    *(uint4*)&As[0][lr][lc] = *(const uint4*)(A + (arow + lr) * E_DIM + k0 + lc);
    *(uint4*)&Bs[0][lr][lc] = *(const uint4*)(W + ((size_t)n0 + lr) * E_DIM + k0 + lc);
    __syncthreads();
    v16h a0 = load_frag16(&As[0][mt2 * 16 + nlane][0], hi);
    v16h a1 = load_frag16(&As[0][(mt2 + 2) * 16 + nlane][0], hi);
    v16h bf = load_frag16(&Bs[0][nt * 16 + nlane][0], hi);
    acc0 = wmma16(a0, bf, acc0);
    acc1 = wmma16(a1, bf, acc1);
  }
#endif

  const int n = n0 + nt * 16 + nlane;
  const float bv = bias ? bias[n] : 0.0f;
#pragma unroll
  for (int t = 0; t < 2; ++t) {
    v8f acc = t ? acc1 : acc0;
    const int mt = mt2 + 2 * t;
#pragma unroll
    for (int r = 0; r < 8; ++r) {
      const int m = m0 + mt * 16 + r + 8 * hi;
      float v = acc[r] + bv;
      if (doRelu) v = fmaxf(v, 0.0f);
      if (resid) {
        const size_t rr = (size_t)(m / seqR) * strideR + baseR + (m % seqR);
        v += resid[rr * E_DIM + n];
      }
      if (outMode == 0) {
        out16[(size_t)m * E_DIM + n] = (_Float16)v;
      } else if (outMode == 1) {
        out32[(size_t)m * E_DIM + n] = v;
      } else {
        const int bb = m / seqOut, s2 = m % seqOut;
        const int hh = n >> 6, d = n & 63;
        if (outMode == 2)
          out16[((size_t)(bb * H_DIM + hh) * seqOut + s2) * DH + d] = (_Float16)v;
        else
          out16[((size_t)(bb * H_DIM + hh) * DH + d) * seqOut + s2] = (_Float16)v;
      }
    }
  }
}

// ---------------------------------------------------------------------------
// Attention: one workgroup per (b, h, 16-row q tile). 8 waves; wave w owns key
// chunks c = w*16 + i (16 keys each, 128 chunks total over K=2048).
// Pass 1: masked scores via WMMA, per-row chunk max/sum -> LDS (deterministic
//         two-level softmax statistics).
// Pass 2: recompute scores, p = exp(s-M)/L; atomic-add p/H into attn_weights;
//         ctx += P(16x16,zero-padded to K=32) @ V via WMMA; LDS reduce waves.
// ---------------------------------------------------------------------------
__global__ __launch_bounds__(256)
void attn_kernel(const _Float16* __restrict__ qh, const _Float16* __restrict__ kh,
                 const _Float16* __restrict__ vh, const int* __restrict__ kpm,
                 float* __restrict__ attnW, _Float16* __restrict__ ctx) {
  __shared__ float cmax[16][128];
  __shared__ float csum[16][128];
  __shared__ float rowM[16], rowInvL[16];
  __shared__ __align__(16) _Float16 pst[8][16][16];
  __shared__ float cred[8][16][64];

  const int tid = threadIdx.x;
  const int wave = tid >> 5, lane = tid & 31;
  const int nlane = lane & 15, hi = lane >> 4;

  const int wg = blockIdx.x;
  const int qt = wg & 63;           // Q_SEQ/16 = 64 tiles
  const int h  = (wg >> 6) & 7;
  const int b  = wg >> 9;

  const size_t bh = (size_t)(b * H_DIM + h);
  const _Float16* qp = qh + (bh * Q_SEQ + (size_t)(qt * 16 + nlane)) * DH;
  const v16h a0 = load_frag16(qp, hi);        // dh 0..31
  const v16h a1 = load_frag16(qp + 32, hi);   // dh 32..63
  const _Float16* kbase = kh + bh * K_SEQ * DH;   // [key][dh]
  const _Float16* vbase = vh + bh * DH * K_SEQ;   // [dh][key] (transposed)

  const float scale = 0.125f;  // dh^-0.5 = 1/8

  // ---------------- pass 1: softmax statistics ----------------
  for (int i = 0; i < 16; ++i) {
    const int c = wave * 16 + i;
    const int col0 = c * 16;
    const int col = col0 + nlane;                // this lane's key column
    const _Float16* kp = kbase + (size_t)col * DH;
    if (i + 1 < 16)                              // global_prefetch next chunk
      __builtin_prefetch(kp + 16 * DH, 0, 1);
    v16h b0 = load_frag16(kp, hi);
    v16h b1 = load_frag16(kp + 32, hi);
    v8f z = {};
    v8f s = wmma16(a0, b0, z);
    s = wmma16(a1, b1, s);
    const int pad = kpm[b * K_SEQ + col];
    float vals[8];
#pragma unroll
    for (int r = 0; r < 8; ++r) {
      const int m = r + 8 * hi;
      const int dpos = (K_SEQ - Q_SEQ) + qt * 16 + m;  // diagonal key position
      const bool masked = (col > dpos) || (pad != 0 && col != dpos);
      vals[r] = masked ? -3.0e38f : s[r] * scale;
    }
#pragma unroll
    for (int r = 0; r < 8; ++r) {
      float mx = vals[r];
#pragma unroll
      for (int off = 1; off < 16; off <<= 1)
        mx = fmaxf(mx, __shfl_xor(mx, off, 32));
      float ex = __expf(vals[r] - mx);
#pragma unroll
      for (int off = 1; off < 16; off <<= 1)
        ex += __shfl_xor(ex, off, 32);
      if (nlane == 0) {
        cmax[r + 8 * hi][c] = mx;
        csum[r + 8 * hi][c] = ex;
      }
    }
  }
  __syncthreads();
  if (tid < 16) {
    float M = -3.0e38f;
    for (int c = 0; c < 128; ++c) M = fmaxf(M, cmax[tid][c]);
    float L = 0.0f;
    for (int c = 0; c < 128; ++c) L += __expf(cmax[tid][c] - M) * csum[tid][c];
    rowM[tid] = M;
    rowInvL[tid] = (L > 0.0f) ? (1.0f / L) : 0.0f;
  }
  __syncthreads();

  // ---------------- pass 2: probs + context ----------------
  v8f ctxacc[4] = {};
  for (int i = 0; i < 16; ++i) {
    const int c = wave * 16 + i;
    const int col0 = c * 16;
    const int col = col0 + nlane;
    const _Float16* kp = kbase + (size_t)col * DH;
    if (i + 1 < 16) {
      __builtin_prefetch(kp + 16 * DH, 0, 1);
      __builtin_prefetch(vbase + (size_t)nlane * K_SEQ + col0 + 16, 0, 1);
    }
    v16h b0 = load_frag16(kp, hi);
    v16h b1 = load_frag16(kp + 32, hi);
    v8f z = {};
    v8f s = wmma16(a0, b0, z);
    s = wmma16(a1, b1, s);
    const int pad = kpm[b * K_SEQ + col];
#pragma unroll
    for (int r = 0; r < 8; ++r) {
      const int m = r + 8 * hi;
      const int dpos = (K_SEQ - Q_SEQ) + qt * 16 + m;
      const bool masked = (col > dpos) || (pad != 0 && col != dpos);
      const float val = masked ? -3.0e38f : s[r] * scale;
      const float p = __expf(val - rowM[m]) * rowInvL[m];
      atomicAdd(&attnW[((size_t)(b * Q_SEQ + qt * 16 + m)) * K_SEQ + col],
                p * 0.125f);  // mean over H=8 heads
      pst[wave][m][nlane] = (_Float16)p;
    }
    // P as A-fragment: rows = q (nlane), k-elems 0..15 = keys, 16..31 = 0 pad
    FragU pa;
    pa.u[0] = *(const uint4*)&pst[wave][nlane][hi * 8];
    pa.u[1] = make_uint4(0u, 0u, 0u, 0u);
#pragma unroll
    for (int j = 0; j < 4; ++j) {   // dh in 4 tiles of 16
      const _Float16* vp = vbase + (size_t)(j * 16 + nlane) * K_SEQ + col0;
      FragU bv;
      bv.u[0] = *(const uint4*)(vp + hi * 8);
      bv.u[1] = make_uint4(0u, 0u, 0u, 0u);
      ctxacc[j] = wmma16(pa.v, bv.v, ctxacc[j]);
    }
  }

  // reduce partial ctx over the 8 waves
#pragma unroll
  for (int j = 0; j < 4; ++j)
#pragma unroll
    for (int r = 0; r < 8; ++r)
      cred[wave][r + 8 * hi][j * 16 + nlane] = ctxacc[j][r];
  __syncthreads();
  for (int idx = tid; idx < 16 * 64; idx += 256) {
    const int m = idx >> 6, d = idx & 63;
    float sum = 0.0f;
#pragma unroll
    for (int w = 0; w < 8; ++w) sum += cred[w][m][d];
    ctx[((size_t)(b * Q_SEQ + qt * 16 + m)) * E_DIM + h * DH + d] = (_Float16)sum;
  }
}

// ---------------------------------------------------------------------------
// LayerNorm over E=512: one block per row, 256 threads x 2 elements.
// ---------------------------------------------------------------------------
__global__ __launch_bounds__(256)
void layernorm_k(const float* __restrict__ in, const float* __restrict__ g,
                 const float* __restrict__ bta, float* __restrict__ out32,
                 _Float16* __restrict__ out16) {
  __shared__ float red[16];
  const int row = blockIdx.x, tid = threadIdx.x;
  const size_t base = (size_t)row * E_DIM;
  const float x0 = in[base + tid];
  const float x1 = in[base + tid + 256];
  float s = x0 + x1, sq = x0 * x0 + x1 * x1;
#pragma unroll
  for (int off = 1; off < 32; off <<= 1) {
    s += __shfl_xor(s, off, 32);
    sq += __shfl_xor(sq, off, 32);
  }
  const int wave = tid >> 5;
  if ((tid & 31) == 0) { red[wave] = s; red[wave + 8] = sq; }
  __syncthreads();
  if (tid == 0) {
    float S = 0.0f, SQ = 0.0f;
    for (int w = 0; w < 8; ++w) { S += red[w]; SQ += red[w + 8]; }
    const float mean = S / (float)E_DIM;
    red[0] = mean;
    red[1] = SQ / (float)E_DIM - mean * mean;
  }
  __syncthreads();
  const float mean = red[0];
  const float rstd = __frsqrt_rn(red[1] + 1e-5f);
  const float y0 = (x0 - mean) * rstd * g[tid] + bta[tid];
  const float y1 = (x1 - mean) * rstd * g[tid + 256] + bta[tid + 256];
  out32[base + tid] = y0;
  out32[base + tid + 256] = y1;
  if (out16) {
    out16[base + tid] = (_Float16)y0;
    out16[base + tid + 256] = (_Float16)y1;
  }
}

// ---------------------------------------------------------------------------
// Host-side launch
// ---------------------------------------------------------------------------
extern "C" void kernel_launch(void* const* d_in, const int* in_sizes, int n_in,
                              void* d_out, int out_size, void* d_ws, size_t ws_size,
                              hipStream_t stream) {
  (void)in_sizes; (void)n_in; (void)out_size; (void)ws_size;

  const float* x_key     = (const float*)d_in[0];
  const float* in_proj_w = (const float*)d_in[1];
  const float* in_proj_b = (const float*)d_in[2];
  const float* out_w     = (const float*)d_in[3];
  const float* out_b     = (const float*)d_in[4];
  const float* ln1_g     = (const float*)d_in[5];
  const float* ln1_b     = (const float*)d_in[6];
  const float* w1        = (const float*)d_in[7];
  const float* b1        = (const float*)d_in[8];
  const float* w2        = (const float*)d_in[9];
  const float* b2        = (const float*)d_in[10];
  const float* ln2_g     = (const float*)d_in[11];
  const float* ln2_b     = (const float*)d_in[12];
  const int*   kpm       = (const int*)d_in[14];

  const long long NX   = (long long)B_DIM * K_SEQ * E_DIM;   // 16,777,216
  const long long NQ   = (long long)B_DIM * Q_SEQ * E_DIM;   //  8,388,608
  const long long NATT = (long long)B_DIM * Q_SEQ * K_SEQ;   // 33,554,432
  const size_t EE = (size_t)E_DIM * E_DIM;

  // ---- workspace carve-up (256B aligned) ----
  char* ws = (char*)d_ws;
  size_t off = 0;
  auto alloc = [&](size_t bytes) {
    char* p = ws + off;
    off += (bytes + 255) & ~(size_t)255;
    return p;
  };
  _Float16* x_h  = (_Float16*)alloc((size_t)NX * 2);   // also reused as ctx
  _Float16* w_h  = (_Float16*)alloc(6 * EE * 2);       // wq|wk|wv|out|w1|w2
  _Float16* q_h  = (_Float16*)alloc((size_t)NQ * 2);   // [b,h,q,dh]
  _Float16* k_h  = (_Float16*)alloc((size_t)NX * 2);   // [b,h,k,dh]
  _Float16* v_h  = (_Float16*)alloc((size_t)NX * 2);   // [b,h,dh,k]
  float*    t_raw = (float*)alloc((size_t)NQ * 4);     // pre-LN1 / pre-LN2
  float*    t32   = (float*)alloc((size_t)NQ * 4);     // post-LN1 f32
  _Float16* t_h   = (_Float16*)alloc((size_t)NQ * 2);  // post-LN1 f16
  _Float16* ff_h  = (_Float16*)alloc((size_t)NQ * 2);  // relu(t@w1^T+b1) f16
  _Float16* ctx_h = x_h;                               // reuse (x_h dead)
  float*    f_raw = t_raw;                             // reuse (t_raw dead)

  float* out_f32 = (float*)d_out;            // [B,Q,E]
  float* attnW   = out_f32 + NQ;             // [B,Q,K]

  // ---- 1) casts to f16 ----
  cast_f32_to_f16<<<4096, 256, 0, stream>>>(x_key, x_h, NX);
  cast_f32_to_f16<<<1024, 256, 0, stream>>>(in_proj_w, w_h, (long long)(3 * EE));
  cast_f32_to_f16<<<512, 256, 0, stream>>>(out_w, w_h + 3 * EE, (long long)EE);
  cast_f32_to_f16<<<512, 256, 0, stream>>>(w1, w_h + 4 * EE, (long long)EE);
  cast_f32_to_f16<<<512, 256, 0, stream>>>(w2, w_h + 5 * EE, (long long)EE);

  // ---- 2) zero attention-weights output (accumulated via atomics) ----
  zero_f32_kernel<<<8192, 256, 0, stream>>>(attnW, NATT);

  const int MK = B_DIM * K_SEQ;  // 32768
  const int MQ = B_DIM * Q_SEQ;  // 16384

  // ---- 3) K = x @ wk^T + bk  -> [b,h,k,dh] f16 ----
  gemm_wmma_nt<<<dim3(MK / 64, E_DIM / 64), 256, 0, stream>>>(
      x_h, w_h + 1 * EE, in_proj_b + E_DIM, nullptr, nullptr, k_h,
      MK, MK, 0, 1, 0, 0, /*mode*/ 2, K_SEQ, 0);
  // ---- 4) V = x @ wv^T + bv  -> [b,h,dh,k] f16 (transposed) ----
  gemm_wmma_nt<<<dim3(MK / 64, E_DIM / 64), 256, 0, stream>>>(
      x_h, w_h + 2 * EE, in_proj_b + 2 * E_DIM, nullptr, nullptr, v_h,
      MK, MK, 0, 1, 0, 0, /*mode*/ 3, K_SEQ, 0);
  // ---- 5) Q = x[:, -Q:] @ wq^T + bq -> [b,h,q,dh] f16 (gathered rows) ----
  gemm_wmma_nt<<<dim3(MQ / 64, E_DIM / 64), 256, 0, stream>>>(
      x_h, w_h, in_proj_b, nullptr, nullptr, q_h,
      Q_SEQ, K_SEQ, K_SEQ - Q_SEQ, 1, 0, 0, /*mode*/ 2, Q_SEQ, 0);

  // ---- 6) attention ----
  attn_kernel<<<B_DIM * H_DIM * (Q_SEQ / 16), 256, 0, stream>>>(
      q_h, k_h, v_h, kpm, attnW, ctx_h);

  // ---- 7) out-proj + residual(query): t_raw = ctx@out_w^T + out_b + x[:, -Q:]
  gemm_wmma_nt<<<dim3(MQ / 64, E_DIM / 64), 256, 0, stream>>>(
      ctx_h, w_h + 3 * EE, out_b, x_key, t_raw, nullptr,
      MQ, MQ, 0, Q_SEQ, K_SEQ, K_SEQ - Q_SEQ, /*mode*/ 1, Q_SEQ, 0);

  // ---- 8) LN1 -> t32 (f32) + t_h (f16) ----
  layernorm_k<<<MQ, 256, 0, stream>>>(t_raw, ln1_g, ln1_b, t32, t_h);

  // ---- 9) FFN1: ff = relu(t @ w1^T + b1) -> f16 ----
  gemm_wmma_nt<<<dim3(MQ / 64, E_DIM / 64), 256, 0, stream>>>(
      t_h, w_h + 4 * EE, b1, nullptr, nullptr, ff_h,
      MQ, MQ, 0, 1, 0, 0, /*mode*/ 0, Q_SEQ, 1);

  // ---- 10) FFN2 + residual(t): f_raw = ff @ w2^T + b2 + t ----
  gemm_wmma_nt<<<dim3(MQ / 64, E_DIM / 64), 256, 0, stream>>>(
      ff_h, w_h + 5 * EE, b2, t32, f_raw, nullptr,
      MQ, MQ, 0, MQ, MQ, 0, /*mode*/ 1, Q_SEQ, 0);

  // ---- 11) LN2 -> final output ----
  layernorm_k<<<MQ, 256, 0, stream>>>(f_raw, ln2_g, ln2_b, out_f32, nullptr);
}
